// res_GAT_62199716380999
// MI455X (gfx1250) — matrix operand order repaired and checked by reading, not compiled, not run on hardware
//
#include <hip/hip_runtime.h>
#include <hip/hip_bf16.h>

typedef __attribute__((ext_vector_type(16))) _Float16 v16h;
typedef __attribute__((ext_vector_type(8)))  float    v8f;
typedef __attribute__((ext_vector_type(4)))  float    f32x4;

#define B_      4
#define N_      20000
#define K_      10
#define F_      128        // 2*C_IN
#define COUT_   128
#define HEADS_  32
#define TILE_N  16
#define NWAVES  10
#define NTHREADS (NWAVES * 32)

// DPP8 selector: lane i (within group of 8) reads lane sel[i]
#define DPP8SEL(a,b,c,d,e,f,g,h) \
    ((a)|((b)<<3)|((c)<<6)|((d)<<9)|((e)<<12)|((f)<<15)|((g)<<18)|((h)<<21))

// Sum across the 4 lanes of a quad (lanes 4q..4q+3) using pure-VALU DPP8 swaps.
__device__ __forceinline__ float quad_sum(float p) {
    int   i1 = __builtin_amdgcn_mov_dpp8(__builtin_bit_cast(int, p),
                                         DPP8SEL(1,0,3,2,5,4,7,6));   // xor 1
    float s1 = p + __builtin_bit_cast(float, i1);
    int   i2 = __builtin_amdgcn_mov_dpp8(__builtin_bit_cast(int, s1),
                                         DPP8SEL(2,3,0,1,6,7,4,5));   // xor 2
    return s1 + __builtin_bit_cast(float, i2);
}

// ---------------------------------------------------------------------------
// Prep: write weights (and mlp1@mlp2) as f16 *pre-swizzled into WMMA B-fragment
// lane order*: frag[tile][kstep] -> lane L holds 16 contiguous halves
//   (column c = tile*16 + L%16, K = kstep*32 + (L/16)*8 + e%8 + (e/8)*16 for half e).
// A wave's fragment load is then one coalesced 1KB burst.
// ---------------------------------------------------------------------------
__global__ void prep_kernel(const float* __restrict__ weights,
                            const float* __restrict__ mlp1,
                            const float* __restrict__ mlp2,
                            _Float16* __restrict__ bswz,   // [8][4][32][16] halves
                            _Float16* __restrict__ bresz)  // [8][2][32][16] halves
{
    const int tid = threadIdx.x;
    for (int idx = tid; idx < 8 * 4 * 32 * 16; idx += blockDim.x) {
        int e = idx & 15;
        int L = (idx >> 4) & 31;
        int s = (idx >> 9) & 3;
        int j = idx >> 11;
        int c  = j * 16 + (L & 15);
        int kk = s * 32 + ((L >> 4) << 3) + (e & 7) + ((e >> 3) << 4);
        bswz[idx] = (_Float16)weights[kk * COUT_ + c];
    }
    for (int idx = tid; idx < 8 * 2 * 32 * 16; idx += blockDim.x) {
        int e = idx & 15;
        int L = (idx >> 4) & 31;
        int s = (idx >> 9) & 1;
        int ct = idx >> 10;
        int c  = ct * 16 + (L & 15);
        int kk = s * 32 + ((L >> 4) << 3) + (e & 7) + ((e >> 3) << 4); // < 64
        float sum = 0.f;                                    // Wm = mlp1 @ mlp2
        for (int cc = 0; cc < COUT_; ++cc)
            sum += mlp1[kk * COUT_ + cc] * mlp2[cc * COUT_ + c];
        bresz[idx] = (_Float16)sum;
    }
}

// ---------------------------------------------------------------------------
// Main fused GAT kernel: one block per (b, 16-row n tile); wave w owns k = w.
// ---------------------------------------------------------------------------
__global__ __launch_bounds__(NTHREADS)
void gat_kernel(const float* __restrict__ x,        // [B][N][K][128]
                const float* __restrict__ a,        // [4][32]
                const v16h*  __restrict__ bswz,     // swizzled weights frags
                const v16h*  __restrict__ bresz,    // swizzled residual frags
                float* __restrict__ out)            // [B][128][N]
{
    __shared__ float out_lds[TILE_N][COUT_];          // 8 KB accumulator
    __shared__ float ahw_lds[2][K_][TILE_N][4];       // double-buffered logits
    __shared__ float mx_lds[2][TILE_N][4];
    __shared__ float rs_lds[2][TILE_N][4];
    __shared__ float a_lds[4][HEADS_];

    const int tid  = threadIdx.x;
    const int lane = tid & 31;
    const int w    = tid >> 5;        // wave id == k slice
    const int hi   = lane >> 4;       // 0/1 half-wave
    const int lo   = lane & 15;

    const int b  = blockIdx.x / (N_ / TILE_N);
    const int n0 = (blockIdx.x % (N_ / TILE_N)) * TILE_N;

    for (int i = tid; i < TILE_N * COUT_; i += NTHREADS)
        out_lds[i >> 7][i & 127] = 0.f;
    for (int i = tid; i < 4 * HEADS_; i += NTHREADS)
        a_lds[i >> 5][i & 31] = a[i];
    __syncthreads();

    // ---- A fragments: x[b, n0+lo, k=w, :128] f32 -> f16, held for all ntiles.
    // ISA 16-bit A 16x32 layout: halves 0..7 <- K = hi*8 + 0..7; 8..15 <- +16.
    // x is a 410MB read-once stream: non-temporal so it doesn't churn L2.
    const int k = w;
    const float* xrow = x + ((size_t)(b * N_ + n0 + lo) * K_ + k) * F_;
    v16h afrag[4];
    #pragma unroll
    for (int s = 0; s < 4; ++s) {
        const float* p = xrow + s * 32 + hi * 8;
        f32x4 q0 = __builtin_nontemporal_load((const f32x4*)(p));
        f32x4 q1 = __builtin_nontemporal_load((const f32x4*)(p + 4));
        f32x4 q2 = __builtin_nontemporal_load((const f32x4*)(p + 16));
        f32x4 q3 = __builtin_nontemporal_load((const f32x4*)(p + 20));
        #pragma unroll
        for (int e = 0; e < 4; ++e) {
            afrag[s][e]      = (_Float16)q0[e];
            afrag[s][4 + e]  = (_Float16)q1[e];
            afrag[s][8 + e]  = (_Float16)q2[e];
            afrag[s][12 + e] = (_Float16)q3[e];
        }
    }

    const int hl = lo >> 2;                // head within the 16-wide c tile

    for (int j = 0; j < COUT_ / 16; ++j) { // 8 output-channel tiles
        const int pb = j & 1;              // LDS double-buffer parity
        const int c  = j * 16 + lo;        // this lane's channel (C/D: lane = col)

        // ---- hoist all 4 B fragments (coalesced, L2-hot), then 4 WMMAs b2b
        v16h bf[4];
        #pragma unroll
        for (int s = 0; s < 4; ++s)
            bf[s] = bswz[(j * 4 + s) * 32 + lane];
        v8f acc = {};
        #pragma unroll
        for (int s = 0; s < 4; ++s)
            acc = __builtin_amdgcn_wmma_f32_16x16x32_f16(
                false, afrag[s], false, bf[s], (short)0, acc, false, false);

        // ---- ahw[n,h] = sum_d leaky(hw[n,4h+d]) * a[d][h]  (VALU quad reduce)
        const float ad = a_lds[c & 3][c >> 2];
        float ahw[8];
        #pragma unroll
        for (int v = 0; v < 8; ++v) {
            float hwv = acc[v];
            ahw[v] = quad_sum((hwv > 0.f ? hwv : 0.2f * hwv) * ad);
        }
        if ((lane & 3) == 0) {
            #pragma unroll
            for (int v = 0; v < 8; ++v)
                ahw_lds[pb][k][v + 8 * hi][hl] = ahw[v];
        }
        __syncthreads();
        // ---- softmax denominators over k (64 (n,h) pairs)
        if (tid < TILE_N * 4) {
            const int nn = tid >> 2, hh = tid & 3;
            float mx = -3.4e38f;
            #pragma unroll
            for (int kk = 0; kk < K_; ++kk)
                mx = fmaxf(mx, ahw_lds[pb][kk][nn][hh]);
            float ssum = 0.f;
            #pragma unroll
            for (int kk = 0; kk < K_; ++kk)
                ssum += __expf(ahw_lds[pb][kk][nn][hh] - mx);
            mx_lds[pb][nn][hh] = mx;
            rs_lds[pb][nn][hh] = 1.0f / ssum;
        }
        __syncthreads();
        // ---- out[n,c] += attn(k,n,h) * hw[n,c] (accumulate across 10 waves)
        #pragma unroll
        for (int v = 0; v < 8; ++v) {
            const int nn = v + 8 * hi;
            float attn = __expf(ahw[v] - mx_lds[pb][nn][hl]) * rs_lds[pb][nn][hl];
            atomicAdd(&out_lds[nn][c], attn * acc[v]);
        }
        // no trailing barrier: next iteration uses the other LDS buffer
    }

    // ---- residual: x[b, n, 0, :64] @ (mlp1@mlp2); waves 0..7 own ctile = w
    if (w < 8) {
        const int c = w * 16 + lo;
        const float* xr = x + ((size_t)(b * N_ + n0 + lo) * K_) * F_;  // k = 0
        v8f acc = {};
        #pragma unroll
        for (int s = 0; s < 2; ++s) {
            const float* p = xr + s * 32 + hi * 8;
            f32x4 q0 = __builtin_nontemporal_load((const f32x4*)(p));
            f32x4 q1 = __builtin_nontemporal_load((const f32x4*)(p + 4));
            f32x4 q2 = __builtin_nontemporal_load((const f32x4*)(p + 16));
            f32x4 q3 = __builtin_nontemporal_load((const f32x4*)(p + 20));
            v16h af;
            #pragma unroll
            for (int e = 0; e < 4; ++e) {
                af[e]      = (_Float16)q0[e];
                af[4 + e]  = (_Float16)q1[e];
                af[8 + e]  = (_Float16)q2[e];
                af[12 + e] = (_Float16)q3[e];
            }
            v16h bfr = bresz[(w * 2 + s) * 32 + lane];
            acc = __builtin_amdgcn_wmma_f32_16x16x32_f16(
                false, af, false, bfr, (short)0, acc, false, false);
        }
        #pragma unroll
        for (int v = 0; v < 8; ++v)
            atomicAdd(&out_lds[v + 8 * hi][c], acc[v]);
    }
    __syncthreads();

    // ---- write transposed output: out[b, c, n0+n]; write-once -> non-temporal
    for (int i = tid; i < TILE_N * COUT_; i += NTHREADS) {
        const int cc = i >> 4, nn = i & 15;
        __builtin_nontemporal_store(out_lds[nn][cc],
            out + ((size_t)b * COUT_ + cc) * N_ + n0 + nn);
    }
}

// ---------------------------------------------------------------------------
extern "C" void kernel_launch(void* const* d_in, const int* in_sizes, int n_in,
                              void* d_out, int out_size, void* d_ws, size_t ws_size,
                              hipStream_t stream)
{
    const float* x       = (const float*)d_in[0];
    const float* weights = (const float*)d_in[1];
    const float* a       = (const float*)d_in[2];
    const float* mlp1    = (const float*)d_in[3];
    const float* mlp2    = (const float*)d_in[4];
    float* out = (float*)d_out;

    _Float16* bswz  = (_Float16*)d_ws;               // 8*4*32*16 halves = 32 KB
    _Float16* bresz = bswz + 8 * 4 * 32 * 16;        // 8*2*32*16 halves = 16 KB

    prep_kernel<<<1, 256, 0, stream>>>(weights, mlp1, mlp2, bswz, bresz);
    gat_kernel<<<B_ * (N_ / TILE_N), NTHREADS, 0, stream>>>(
        x, a, (const v16h*)bswz, (const v16h*)bresz, out);
}
